// GPT2_56418690401009
// MI455X (gfx1250) — compile-verified, matrix operand order checked by reading, not run on hardware
//
#include <hip/hip_runtime.h>
#include <hip/hip_bf16.h>
#include <math.h>
#include <stdint.h>

#define GPT_D   768
#define GPT_H   12
#define GPT_HD  64
#define GPT_T   1024
#define GPT_B   4
#define GPT_L   12
#define GPT_V   50257
#define GPT_M   (GPT_B * GPT_T)   /* 4096 token rows */

typedef __bf16 v16bf  __attribute__((ext_vector_type(16)));
typedef __bf16 bf16x8 __attribute__((ext_vector_type(8)));
typedef __bf16 bf16x4 __attribute__((ext_vector_type(4)));
typedef float  v8f    __attribute__((ext_vector_type(8)));
typedef unsigned int u32x4 __attribute__((ext_vector_type(4)));
typedef int          i32x4 __attribute__((ext_vector_type(4)));
typedef int          i32x8 __attribute__((ext_vector_type(8)));

// Tensor Data Mover availability (device pass only; host pass falls back)
#if defined(__has_builtin)
#if __has_builtin(__builtin_amdgcn_tensor_load_to_lds)
#define HAVE_TDM 1
#endif
#endif
#ifndef HAVE_TDM
#define HAVE_TDM 0
#endif

// K-index map for 16-bit WMMA fragments (ISA 7.12.2):
// element e (0..15) of the packed v16bf, hl = lane>>4 (upper half-wave gets +8)
__device__ __forceinline__ int kmap(int e, int hl) {
  return ((e < 8) ? e : (e + 8)) + hl * 8;
}

__device__ __forceinline__ float gelu_tanh(float x) {
  float x3 = x * x * x;
  return 0.5f * x * (1.0f + tanhf(0.7978845608028654f * (x + 0.044715f * x3)));
}

// Build a 16-element bf16 WMMA fragment from a padded-row LDS tile with two
// aligned 16-byte ds_load_b128s (K runs: [hl*8, hl*8+7] and [16+hl*8, 23+hl*8]).
__device__ __forceinline__ v16bf frag_from_row(const __bf16* rowp, int hl) {
  const bf16x8 lo = *reinterpret_cast<const bf16x8*>(rowp + hl * 8);
  const bf16x8 hi = *reinterpret_cast<const bf16x8*>(rowp + 16 + hl * 8);
  return __builtin_shufflevector(lo, hi, 0, 1, 2, 3, 4, 5, 6, 7,
                                 8, 9, 10, 11, 12, 13, 14, 15);
}

#if HAVE_TDM
// Issue a 2-D TDM load: tile_d1 rows of tile_d0 f32 elements, row stride
// `stride0` elements, packed contiguously into LDS at byte offset lds_off.
// D# layout per CDNA5 ISA 8.3/8.4 (group0: count/lds/global/type,
// group1: data_size + dims/tile/strides). Groups 2/3 zero (2-D tile).
__device__ __forceinline__ void tdm_load_2d(uint64_t gaddr, unsigned lds_off,
                                            unsigned tile_d0, unsigned tile_d1,
                                            unsigned tdim0, unsigned tdim1,
                                            unsigned stride0) {
  u32x4 g0;
  g0[0] = 1u;                                               // count=1 (valid), user desc
  g0[1] = lds_off;                                          // lds_addr
  g0[2] = (unsigned)(gaddr & 0xffffffffu);                  // global_addr[31:0]
  g0[3] = (unsigned)((gaddr >> 32) & 0x01ffffffu) | (2u << 30); // ga[56:32], type=2
  i32x8 g1;
  g1[0] = (int)(2u << 16);                                  // data_size=2 (4 bytes)
  g1[1] = (int)((tdim0 & 0xffffu) << 16);                   // tensor_dim0[15:0]
  g1[2] = (int)((tdim0 >> 16) | ((tdim1 & 0xffffu) << 16)); // dim0 hi | dim1 lo
  g1[3] = (int)((tdim1 >> 16) | (tile_d0 << 16));           // dim1 hi | tile_dim0
  g1[4] = (int)(tile_d1 & 0xffffu);                         // tile_dim1 (tile_dim2=0)
  g1[5] = (int)stride0;                                     // tensor_dim0_stride[31:0]
  g1[6] = 0;                                                // stride0 hi | stride1 lo
  g1[7] = 0;                                                // stride1 hi
  i32x4 gz = {0, 0, 0, 0};
#if __clang_major__ >= 23
  i32x8 gz8 = {0, 0, 0, 0, 0, 0, 0, 0};
  __builtin_amdgcn_tensor_load_to_lds(g0, g1, gz, gz, gz8, 0);
#else
  __builtin_amdgcn_tensor_load_to_lds(g0, g1, gz, gz, 0);
#endif
}
#endif

// ---------------------------------------------------------------------------
// Embedding: x[b,t,:] = wte[token] + wpe[t]
// ---------------------------------------------------------------------------
__global__ __launch_bounds__(256)
void gpt2_embed_kernel(const int* __restrict__ tokens,
                       const float* __restrict__ wte,
                       const float* __restrict__ wpe,
                       float* __restrict__ x) {
  const int row = blockIdx.x;              // b*T + t
  const int t   = row % GPT_T;
  const int tok = tokens[row];
  const float* we = wte + (size_t)tok * GPT_D;
  const float* pe = wpe + (size_t)t   * GPT_D;
  float* xr = x + (size_t)row * GPT_D;
#pragma unroll
  for (int i = 0; i < GPT_D / 256; ++i) {
    int c = threadIdx.x + i * 256;
    xr[c] = we[c] + pe[c];
  }
}

// ---------------------------------------------------------------------------
// LayerNorm: one block per row of D=768
// ---------------------------------------------------------------------------
__global__ __launch_bounds__(256)
void gpt2_layernorm_kernel(const float* __restrict__ x,
                           const float* __restrict__ w,
                           const float* __restrict__ b,
                           float* __restrict__ y) {
  __shared__ float sh[256];
  const int tid = threadIdx.x;
  const float* xr = x + (size_t)blockIdx.x * GPT_D;
  float v0 = xr[tid], v1 = xr[tid + 256], v2 = xr[tid + 512];
  float s  = v0 + v1 + v2;
  float ss = v0 * v0 + v1 * v1 + v2 * v2;

  sh[tid] = s; __syncthreads();
  for (int st = 128; st > 0; st >>= 1) {
    if (tid < st) sh[tid] += sh[tid + st];
    __syncthreads();
  }
  float mean = sh[0] * (1.0f / GPT_D);
  __syncthreads();
  sh[tid] = ss; __syncthreads();
  for (int st = 128; st > 0; st >>= 1) {
    if (tid < st) sh[tid] += sh[tid + st];
    __syncthreads();
  }
  float var  = sh[0] * (1.0f / GPT_D) - mean * mean;
  float rstd = rsqrtf(var + 1e-5f);

  float* yr = y + (size_t)blockIdx.x * GPT_D;
  yr[tid]       = (v0 - mean) * rstd * w[tid]       + b[tid];
  yr[tid + 256] = (v1 - mean) * rstd * w[tid + 256] + b[tid + 256];
  yr[tid + 512] = (v2 - mean) * rstd * w[tid + 512] + b[tid + 512];
}

// ---------------------------------------------------------------------------
// bf16 WMMA GEMM:  C[M,N] = act( A[M,K] @ W + bias ) (+ resid)
//   TRANS_B=0: W is [K,N] row-major (ldb = N)
//   TRANS_B=1: W is [N,K] row-major (ldb = K)  -> used for x @ wte^T
// Block tile 128x64, BK=32, 8 waves (32x32 each = 2x2 WMMA tiles).
// Weight tiles are streamed by the Tensor Data Mover into a double-buffered
// f32 LDS stage (TENSORcnt-tracked) and converted to a transposed bf16 tile
// BsT so both A and B fragments are two ds_load_b128 per lane.
// ---------------------------------------------------------------------------
#define AS_BYTES   (128 * 40 * 2)              /* 10240 */
#define BT_OFF     AS_BYTES
#define BT_BYTES   (64 * 40 * 2)               /* 5120  */
#define ST0_OFF    (BT_OFF + BT_BYTES)         /* 15360 */
#define ST_BYTES   (32 * 64 * 4)               /* 8192  */
#define ST1_OFF    (ST0_OFF + ST_BYTES)        /* 23552 */
#define GEMM_SMEM  (ST1_OFF + ST_BYTES)        /* 31744 */

template <bool TRANS_B, bool GELU_ACT, bool RESID>
__global__ __launch_bounds__(256)
void gpt2_gemm_bf16_wmma(const float* __restrict__ A,
                         const float* __restrict__ W,
                         const float* __restrict__ bias,
                         const float* __restrict__ resid,
                         float* __restrict__ C,
                         int M, int N, int K, int lda, int ldb, int ldc) {
  extern __shared__ char gsm[];
  __bf16 (*As)[40]  = reinterpret_cast<__bf16 (*)[40]>(gsm);
  __bf16 (*BsT)[40] = reinterpret_cast<__bf16 (*)[40]>(gsm + BT_OFF);

  const int tid  = threadIdx.x;
  const int lane = tid & 31;
  const int wave = tid >> 5;
  const int wm   = wave & 3;       // wave row  -> *32
  const int wn   = wave >> 2;      // wave col  -> *32
  const int hl   = lane >> 4;
  const int l16  = lane & 15;
  const int mBlock = blockIdx.x * 128;
  const int nBlock = blockIdx.y * 64;
  const int rowLen = (N - nBlock < 64) ? (N - nBlock) : 64;  // valid cols in tile

  v8f acc[2][2];
#pragma unroll
  for (int i = 0; i < 2; ++i)
#pragma unroll
    for (int j = 0; j < 2; ++j)
      acc[i][j] = (v8f){0.f, 0.f, 0.f, 0.f, 0.f, 0.f, 0.f, 0.f};

#if HAVE_TDM
  // Prologue: DMA first weight tile into stage 0 (wave 0 issues; EXEC ignored).
  if (tid < 32) {
    if (!TRANS_B) {
      uint64_t ga = (uint64_t)(uintptr_t)(W + (size_t)0 * ldb + nBlock);
      tdm_load_2d(ga, ST0_OFF, (unsigned)rowLen, 32u,
                  (unsigned)rowLen, 32u, (unsigned)ldb);
    } else {
      uint64_t ga = (uint64_t)(uintptr_t)(W + (size_t)nBlock * ldb + 0);
      tdm_load_2d(ga, ST0_OFF, 32u, (unsigned)rowLen,
                  (unsigned)K, (unsigned)rowLen, (unsigned)ldb);
    }
  }
#endif

  int kt = 0;
  for (int k0 = 0; k0 < K; k0 += 32, ++kt) {
    // ---- A tile: 128x32 f32 -> bf16 in LDS (float4 per thread, 4 rows) ----
    {
      const int r0 = tid >> 3;           // 0..31
      const int c4 = (tid & 7) * 4;      // 0..28
#pragma unroll
      for (int rr = 0; rr < 4; ++rr) {
        int r  = r0 + rr * 32;
        int gr = mBlock + r;
        float4 v = make_float4(0.f, 0.f, 0.f, 0.f);
        if (gr < M)
          v = *reinterpret_cast<const float4*>(A + (size_t)gr * lda + k0 + c4);
        bf16x4 pk = { (__bf16)v.x, (__bf16)v.y, (__bf16)v.z, (__bf16)v.w };
        *reinterpret_cast<bf16x4*>(&As[r][c4]) = pk;
      }
    }
    if (k0 + 32 < K) {
      int gr = mBlock + (tid >> 3);
      if (gr < M) __builtin_prefetch(A + (size_t)gr * lda + k0 + 32, 0, 0);
    }

#if HAVE_TDM
    // ---- publish stage[kt&1] (DMA'd weight tile), then convert to BsT ----
    if (tid < 32) __builtin_amdgcn_s_wait_tensorcnt(0);
    __syncthreads();

    const float* st = reinterpret_cast<const float*>(
        gsm + ((kt & 1) ? ST1_OFF : ST0_OFF));
    if (!TRANS_B) {
      // stage: [32 k][rowLen n] packed ; BsT[n][k]
      const int k  = tid >> 3;           // 0..31
      const int n8 = (tid & 7) * 8;      // 0..56
      if (rowLen == 64) {
        float4 u0 = *reinterpret_cast<const float4*>(st + k * 64 + n8);
        float4 u1 = *reinterpret_cast<const float4*>(st + k * 64 + n8 + 4);
        BsT[n8 + 0][k] = (__bf16)u0.x;  BsT[n8 + 1][k] = (__bf16)u0.y;
        BsT[n8 + 2][k] = (__bf16)u0.z;  BsT[n8 + 3][k] = (__bf16)u0.w;
        BsT[n8 + 4][k] = (__bf16)u1.x;  BsT[n8 + 5][k] = (__bf16)u1.y;
        BsT[n8 + 6][k] = (__bf16)u1.z;  BsT[n8 + 7][k] = (__bf16)u1.w;
      } else {
#pragma unroll
        for (int i = 0; i < 8; ++i) {
          int n = n8 + i;
          float f = (n < rowLen) ? st[k * rowLen + n] : 0.f;
          BsT[n][k] = (__bf16)f;
        }
      }
    } else {
      // stage: [rowLen n][32 k] packed ; BsT[n][k] contiguous copy
      const int n  = tid >> 2;           // 0..63
      const int k8 = (tid & 3) * 8;      // 0..24
      bf16x8 pk;
      if (n < rowLen) {
        float4 u0 = *reinterpret_cast<const float4*>(st + n * 32 + k8);
        float4 u1 = *reinterpret_cast<const float4*>(st + n * 32 + k8 + 4);
        pk = (bf16x8){(__bf16)u0.x, (__bf16)u0.y, (__bf16)u0.z, (__bf16)u0.w,
                      (__bf16)u1.x, (__bf16)u1.y, (__bf16)u1.z, (__bf16)u1.w};
      } else {
        pk = (bf16x8){(__bf16)0.f, (__bf16)0.f, (__bf16)0.f, (__bf16)0.f,
                      (__bf16)0.f, (__bf16)0.f, (__bf16)0.f, (__bf16)0.f};
      }
      *reinterpret_cast<bf16x8*>(&BsT[n][k8]) = pk;
    }

    // ---- kick DMA for the next weight tile into the other stage ----
    if (tid < 32 && k0 + 32 < K) {
      unsigned stOff = ((kt & 1) ^ 1) ? ST1_OFF : ST0_OFF;
      if (!TRANS_B) {
        uint64_t ga = (uint64_t)(uintptr_t)(W + (size_t)(k0 + 32) * ldb + nBlock);
        tdm_load_2d(ga, stOff, (unsigned)rowLen, 32u,
                    (unsigned)rowLen, 32u, (unsigned)ldb);
      } else {
        uint64_t ga = (uint64_t)(uintptr_t)(W + (size_t)nBlock * ldb + (k0 + 32));
        tdm_load_2d(ga, stOff, 32u, (unsigned)rowLen,
                    (unsigned)(K - k0 - 32), (unsigned)rowLen, (unsigned)ldb);
      }
    }
#else
    // ---- fallback: direct global -> BsT (transposed bf16 tile) ----
    if (!TRANS_B) {
      const int k  = tid >> 3;           // 0..31
      const int n8 = (tid & 7) * 8;      // 0..56
#pragma unroll
      for (int i = 0; i < 8; ++i) {
        int gn = nBlock + n8 + i;
        float f = (gn < N) ? W[(size_t)(k0 + k) * ldb + gn] : 0.f;
        BsT[n8 + i][k] = (__bf16)f;
      }
    } else {
      const int n  = tid >> 2;           // 0..63
      const int k8 = (tid & 3) * 8;      // 0..24
      int gn = nBlock + n;
      bf16x8 pk = {(__bf16)0.f, (__bf16)0.f, (__bf16)0.f, (__bf16)0.f,
                   (__bf16)0.f, (__bf16)0.f, (__bf16)0.f, (__bf16)0.f};
      if (gn < N) {
        float4 u0 = *reinterpret_cast<const float4*>(W + (size_t)gn * ldb + k0 + k8);
        float4 u1 = *reinterpret_cast<const float4*>(W + (size_t)gn * ldb + k0 + k8 + 4);
        pk = (bf16x8){(__bf16)u0.x, (__bf16)u0.y, (__bf16)u0.z, (__bf16)u0.w,
                      (__bf16)u1.x, (__bf16)u1.y, (__bf16)u1.z, (__bf16)u1.w};
      }
      *reinterpret_cast<bf16x8*>(&BsT[n][k8]) = pk;
    }
#endif
    __syncthreads();

    // ---- fragments (2x ds_load_b128 each) + 4 WMMAs ----
    v16bf af[2], bw[2];
#pragma unroll
    for (int i = 0; i < 2; ++i)
      af[i] = frag_from_row(&As[wm * 32 + i * 16 + l16][0], hl);
#pragma unroll
    for (int j = 0; j < 2; ++j)
      bw[j] = frag_from_row(&BsT[wn * 32 + j * 16 + l16][0], hl);
#pragma unroll
    for (int i = 0; i < 2; ++i)
#pragma unroll
      for (int j = 0; j < 2; ++j)
        acc[i][j] = __builtin_amdgcn_wmma_f32_16x16x32_bf16(
            false, af[i], false, bw[j], (short)0, acc[i][j], false, false);
    __syncthreads();
  }

  // ---- epilogue: bias -> GELU -> residual ----
#pragma unroll
  for (int i = 0; i < 2; ++i)
#pragma unroll
    for (int j = 0; j < 2; ++j)
#pragma unroll
      for (int r = 0; r < 8; ++r) {
        int gr = mBlock + wm * 32 + i * 16 + r + 8 * hl;
        int gc = nBlock + wn * 32 + j * 16 + l16;
        if (gr < M && gc < N) {
          float v = acc[i][j][r];
          if (bias) v += bias[gc];
          if (GELU_ACT) v = gelu_tanh(v);
          if (RESID) v += resid[(size_t)gr * ldc + gc];
          C[(size_t)gr * ldc + gc] = v;
        }
      }
}

// ---------------------------------------------------------------------------
// Flash attention: one wave per (b, h, 16-query tile).
//   scores = (q/8) @ k^T  (two chained 16x16x32 bf16 WMMAs over HD=64)
//   online causal softmax, P bounced via LDS into A-fragment layout,
//   O += P @ V (four bf16 WMMAs per 32-key block).
// ---------------------------------------------------------------------------
__global__ __launch_bounds__(32)
void gpt2_attention_kernel(const float* __restrict__ qkv,  // [B,T,3D]
                           float* __restrict__ out) {      // [B,T,D]
  __shared__ float pS[16][33];

  const int qtiles = GPT_T / 16;
  const int qt = blockIdx.x % qtiles;
  const int hh = (blockIdx.x / qtiles) % GPT_H;
  const int bb = blockIdx.x / (qtiles * GPT_H);

  const int lane = threadIdx.x & 31;
  const int hl   = lane >> 4;
  const int l16  = lane & 15;
  const int S3D  = 3 * GPT_D;

  const float* qp = qkv + (size_t)bb * GPT_T * S3D + hh * GPT_HD;             // Q
  const float* kp = qp + GPT_D;                                               // K
  const float* vp = qp + 2 * GPT_D;                                           // V

  // Q fragments (scaled by 1/sqrt(HD) = 1/8), A-layout: row = l16, K = head dim
  v16bf aq[2];
#pragma unroll
  for (int f = 0; f < 2; ++f) {
    int t = qt * 16 + l16;
#pragma unroll
    for (int e = 0; e < 16; ++e) {
      int d = f * 32 + kmap(e, hl);
      aq[f][e] = (__bf16)(qp[(size_t)t * S3D + d] * 0.125f);
    }
  }

  v8f o[4];
#pragma unroll
  for (int c = 0; c < 4; ++c) o[c] = (v8f){0.f,0.f,0.f,0.f,0.f,0.f,0.f,0.f};
  float mrow[8], lrow[8];
#pragma unroll
  for (int r = 0; r < 8; ++r) { mrow[r] = -3.0e38f; lrow[r] = 0.f; }

  const int kbEnd = (qt * 16 + 15) / 32;    // inclusive
  for (int kb = 0; kb <= kbEnd; ++kb) {
    const int k0 = kb * 32;

    // scores for two 16-key column tiles
    v8f s[2];
#pragma unroll
    for (int c = 0; c < 2; ++c) {
      v16bf bk0, bk1;
      int kc = k0 + c * 16 + l16;           // key index = B column
#pragma unroll
      for (int e = 0; e < 16; ++e) {
        int d0 = kmap(e, hl);
        bk0[e] = (__bf16)kp[(size_t)kc * S3D + d0];
        bk1[e] = (__bf16)kp[(size_t)kc * S3D + 32 + d0];
      }
      v8f z = (v8f){0.f,0.f,0.f,0.f,0.f,0.f,0.f,0.f};
      z = __builtin_amdgcn_wmma_f32_16x16x32_bf16(false, aq[0], false, bk0, (short)0, z, false, false);
      s[c] = __builtin_amdgcn_wmma_f32_16x16x32_bf16(false, aq[1], false, bk1, (short)0, z, false, false);
    }

    // causal mask + online softmax (row = r + 8*hl within tile; col across lanes)
#pragma unroll
    for (int c = 0; c < 2; ++c)
#pragma unroll
      for (int r = 0; r < 8; ++r) {
        int row = qt * 16 + r + 8 * hl;
        int col = k0 + c * 16 + l16;
        if (col > row) s[c][r] = -3.0e38f;
      }
#pragma unroll
    for (int r = 0; r < 8; ++r) {
      float mx = fmaxf(s[0][r], s[1][r]);
      for (int off = 8; off > 0; off >>= 1) mx = fmaxf(mx, __shfl_xor(mx, off, 32));
      float nm    = fmaxf(mrow[r], mx);
      float alpha = __expf(mrow[r] - nm);
      mrow[r] = nm;
      s[0][r] = __expf(s[0][r] - nm);
      s[1][r] = __expf(s[1][r] - nm);
      float rs = s[0][r] + s[1][r];
      for (int off = 8; off > 0; off >>= 1) rs += __shfl_xor(rs, off, 32);
      lrow[r] = lrow[r] * alpha + rs;
#pragma unroll
      for (int c = 0; c < 4; ++c) o[c][r] *= alpha;
    }

    // transpose P (C-layout -> A-layout) through LDS
#pragma unroll
    for (int c = 0; c < 2; ++c)
#pragma unroll
      for (int r = 0; r < 8; ++r)
        pS[r + 8 * hl][c * 16 + l16] = s[c][r];
    __syncthreads();
    v16bf ap;
#pragma unroll
    for (int e = 0; e < 16; ++e) ap[e] = (__bf16)pS[l16][kmap(e, hl)];
    __syncthreads();

    // O += P @ V
#pragma unroll
    for (int c = 0; c < 4; ++c) {
      v16bf bv;
      int d = c * 16 + l16;
#pragma unroll
      for (int e = 0; e < 16; ++e) {
        int kr = k0 + kmap(e, hl);
        bv[e] = (__bf16)vp[(size_t)kr * S3D + d];
      }
      o[c] = __builtin_amdgcn_wmma_f32_16x16x32_bf16(false, ap, false, bv, (short)0, o[c], false, false);
    }
  }

  // normalize and write [B,T,D]
#pragma unroll
  for (int c = 0; c < 4; ++c)
#pragma unroll
    for (int r = 0; r < 8; ++r) {
      int row = qt * 16 + r + 8 * hl;
      int d   = c * 16 + l16;
      out[((size_t)bb * GPT_T + row) * GPT_D + hh * GPT_HD + d] = o[c][r] / lrow[r];
    }
}

// ---------------------------------------------------------------------------
// Host-side launch
// ---------------------------------------------------------------------------
template <bool TB, bool G, bool R>
static inline void launch_gemm(const float* A, const float* W, const float* bias,
                               const float* resid, float* C,
                               int M, int N, int K, int ldb, hipStream_t s) {
  dim3 grid((M + 127) / 128, (N + 63) / 64);
  gpt2_gemm_bf16_wmma<TB, G, R><<<grid, 256, GEMM_SMEM, s>>>(A, W, bias, resid, C,
                                                             M, N, K, K, ldb, N);
}

extern "C" void kernel_launch(void* const* d_in, const int* in_sizes, int n_in,
                              void* d_out, int out_size, void* d_ws, size_t ws_size,
                              hipStream_t stream) {
  (void)in_sizes; (void)n_in; (void)out_size; (void)ws_size;

  const int*   tokens = (const int*)  d_in[0];
  const float* wte    = (const float*)d_in[1];
  const float* wpe    = (const float*)d_in[2];
  const float* ln1_w  = (const float*)d_in[3];
  const float* ln1_b  = (const float*)d_in[4];
  const float* qkv_w  = (const float*)d_in[5];
  const float* qkv_b  = (const float*)d_in[6];
  const float* proj_w = (const float*)d_in[7];
  const float* proj_b = (const float*)d_in[8];
  const float* ln2_w  = (const float*)d_in[9];
  const float* ln2_b  = (const float*)d_in[10];
  const float* fc_w   = (const float*)d_in[11];
  const float* fc_b   = (const float*)d_in[12];
  const float* fcp_w  = (const float*)d_in[13];
  const float* fcp_b  = (const float*)d_in[14];
  const float* lnf_w  = (const float*)d_in[15];
  const float* lnf_b  = (const float*)d_in[16];

  float* ws   = (float*)d_ws;
  float* x    = ws;                                    // [M, D]
  float* hbuf = x    + (size_t)GPT_M * GPT_D;          // [M, D]
  float* qkvb = hbuf + (size_t)GPT_M * GPT_D;          // [M, 3D]
  float* att  = qkvb + (size_t)GPT_M * 3 * GPT_D;      // [M, D]
  float* h2   = att  + (size_t)GPT_M * GPT_D;          // [M, 4D]

  gpt2_embed_kernel<<<GPT_M, 256, 0, stream>>>(tokens, wte, wpe, x);

  for (int l = 0; l < GPT_L; ++l) {
    gpt2_layernorm_kernel<<<GPT_M, 256, 0, stream>>>(
        x, ln1_w + (size_t)l * GPT_D, ln1_b + (size_t)l * GPT_D, hbuf);

    launch_gemm<false, false, false>(hbuf, qkv_w + (size_t)l * GPT_D * 3 * GPT_D,
                                     qkv_b + (size_t)l * 3 * GPT_D, nullptr, qkvb,
                                     GPT_M, 3 * GPT_D, GPT_D, 3 * GPT_D, stream);

    gpt2_attention_kernel<<<GPT_B * GPT_H * (GPT_T / 16), 32, 0, stream>>>(qkvb, att);

    launch_gemm<false, false, true>(att, proj_w + (size_t)l * GPT_D * GPT_D,
                                    proj_b + (size_t)l * GPT_D, x, x,
                                    GPT_M, GPT_D, GPT_D, GPT_D, stream);

    gpt2_layernorm_kernel<<<GPT_M, 256, 0, stream>>>(
        x, ln2_w + (size_t)l * GPT_D, ln2_b + (size_t)l * GPT_D, hbuf);

    launch_gemm<false, true, false>(hbuf, fc_w + (size_t)l * GPT_D * 4 * GPT_D,
                                    fc_b + (size_t)l * 4 * GPT_D, nullptr, h2,
                                    GPT_M, 4 * GPT_D, GPT_D, 4 * GPT_D, stream);

    launch_gemm<false, false, true>(h2, fcp_w + (size_t)l * 4 * GPT_D * GPT_D,
                                    fcp_b + (size_t)l * GPT_D, x, x,
                                    GPT_M, GPT_D, 4 * GPT_D, GPT_D, stream);
  }

  gpt2_layernorm_kernel<<<GPT_M, 256, 0, stream>>>(x, lnf_w, lnf_b, hbuf);

  // logits = hbuf @ wte^T  (tied lm head), N = 50257 (bounds-checked)
  launch_gemm<true, false, false>(hbuf, wte, nullptr, nullptr, (float*)d_out,
                                  GPT_M, GPT_V, GPT_D, GPT_D, stream);
}
